// KeyNet_11570641895878
// MI455X (gfx1250) — compile-verified, matrix operand order checked by reading, not run on hardware
//
#include <hip/hip_runtime.h>

// ---------------------------------------------------------------------------
// KeyNet on MI455X (gfx1250): fold anchor-independent embedding branches into
// per-point bias tensors (c5e, c1e), then run the per-anchor GEMM chain with
// v_wmma_f32_16x16x32_f16. Each wave owns one N-column with FOUR independent
// accumulator chains (mt=0..3) sharing one B fragment and double-buffered
// fragment sets; launch_bounds(256,2) since one 8-wave WG per WGP (105 KB LDS)
// only needs 2 resident waves per SIMD.
// ---------------------------------------------------------------------------

typedef _Float16 v8h  __attribute__((ext_vector_type(8)));
typedef _Float16 v16h __attribute__((ext_vector_type(16)));
typedef float    v8f  __attribute__((ext_vector_type(8)));

#define NUM_A 125
#define NUM_P 1024
#define MT    64      // points per M-tile
#define NW    8       // waves per workgroup (256 threads)

// Load a 16-half fragment: two 8-half (16B) runs at +0 and +16 halves,
// matching the ISA 16-bit A/B fragment layout (K split across lane halves).
static __device__ inline v16h frag16(const _Float16* p) {
  v8h lo = *(const v8h*)(p);
  v8h hi = *(const v8h*)(p + 16);
  return __builtin_shufflevector(lo, hi, 0,1,2,3,4,5,6,7,8,9,10,11,12,13,14,15);
}

static __device__ inline v8f wmma16(v16h a, v16h b, v8f c) {
  return __builtin_amdgcn_wmma_f32_16x16x32_f16(false, a, false, b, (short)0, c,
                                                false, false);
}

static __device__ inline float lrelu(float v) { return v >= 0.f ? v : 0.01f * v; }

// Quad-chain K-loop with software-pipelined (double-buffered) fragments.
// aB: LDS activation base for this lane (row lm, K-offset applied).
// bB: global weight base for this lane's N-column.
template <int NK, int STRIDE>
static __device__ inline void gemm4(const _Float16* __restrict__ aB,
                                    const _Float16* __restrict__ bB,
                                    v8f& a0, v8f& a1, v8f& a2, v8f& a3) {
  v16h b  = frag16(bB);
  v16h f0 = frag16(aB);
  v16h f1 = frag16(aB + 16 * STRIDE);
  v16h f2 = frag16(aB + 32 * STRIDE);
  v16h f3 = frag16(aB + 48 * STRIDE);
#pragma unroll
  for (int kt = 0; kt < NK; ++kt) {
    v16h cb = b, c0 = f0, c1 = f1, c2 = f2, c3 = f3;
    if (kt + 1 < NK) {  // issue next-kt loads before consuming current frags
      const _Float16* an = aB + (kt + 1) * 32;
      b  = frag16(bB + (kt + 1) * 32);
      f0 = frag16(an);
      f1 = frag16(an + 16 * STRIDE);
      f2 = frag16(an + 32 * STRIDE);
      f3 = frag16(an + 48 * STRIDE);
    }
    a0 = wmma16(c0, cb, a0);
    a1 = wmma16(c1, cb, a1);
    a2 = wmma16(c2, cb, a2);
    a3 = wmma16(c3, cb, a3);
  }
}

// ---------------------------------------------------------------------------
// Kernel 0: per-point embedding chain + folded bias tensors.
// emb[32] -> e1[64] -> e2[128]; c5e = W5[:,128:256]@e2 + b5;
// c1e = W1[:,64:128]@e1 + W1[:,256:384]@e2 + b1.
// ---------------------------------------------------------------------------
__global__ __launch_bounds__(128) void keynet_point(
    const float* __restrict__ out_img, const int* __restrict__ choose,
    const float* __restrict__ We1, const float* __restrict__ be1,
    const float* __restrict__ We2, const float* __restrict__ be2,
    const float* __restrict__ W5,  const float* __restrict__ b5,
    const float* __restrict__ WA1, const float* __restrict__ bA1,
    float* __restrict__ c5e, float* __restrict__ c1e) {
  __shared__ float emb[32];
  __shared__ float e1[64];
  __shared__ float e2[128];
  const int p = blockIdx.x;
  const int tid = threadIdx.x;
  const int idx = choose[p];

  if (tid < 32) emb[tid] = out_img[tid * 9216 + idx];
  __syncthreads();
  if (tid < 64) {
    float s = be1[tid];
    for (int c = 0; c < 32; ++c) s += We1[tid * 32 + c] * emb[c];
    e1[tid] = fmaxf(s, 0.f);
  }
  __syncthreads();
  if (tid < 128) {
    float s = be2[tid];
    for (int c = 0; c < 64; ++c) s += We2[tid * 64 + c] * e1[c];
    e2[tid] = fmaxf(s, 0.f);
  }
  __syncthreads();
  for (int n = tid; n < 256; n += 128) {
    float s = b5[n];
    for (int c = 0; c < 128; ++c) s += W5[n * 256 + 128 + c] * e2[c];
    c5e[p * 256 + n] = s;
  }
  for (int n = tid; n < 320; n += 128) {
    float s = bA1[n];
    for (int c = 0; c < 64; ++c)  s += WA1[n * 640 + 64 + c]  * e1[c];
    for (int c = 0; c < 128; ++c) s += WA1[n * 640 + 256 + c] * e2[c];
    c1e[p * 320 + n] = s;
  }
}

// ---------------------------------------------------------------------------
// Kernel 1: convert / repack weights to f16 row-major [N][K_reduced].
// ---------------------------------------------------------------------------
__global__ __launch_bounds__(256) void keynet_wconv(
    const float* __restrict__ W2, const float* __restrict__ W5,
    const float* __restrict__ WA1, const float* __restrict__ WA2,
    _Float16* __restrict__ W2h, _Float16* __restrict__ W5h,
    _Float16* __restrict__ W1h, _Float16* __restrict__ W2ah) {
  int i = blockIdx.x * 256 + threadIdx.x;
  if (i < 8192) {                       // conv2: [128][64]
    W2h[i] = (_Float16)W2[i];
  } else if (i < 8192 + 32768) {        // conv5 anchor part: [256][128]
    int j = i - 8192;
    int n = j >> 7, k = j & 127;
    W5h[j] = (_Float16)W5[n * 256 + k];
  } else if (i < 8192 + 32768 + 143360) {  // all_conv1 reduced: [320][448]
    int j = i - (8192 + 32768);
    int n = j / 448, k = j % 448;
    int ko = (k < 64) ? k : (k < 192 ? k + 64 : k + 192);
    W1h[j] = (_Float16)WA1[n * 640 + ko];
  } else if (i < 8192 + 32768 + 143360 + 51200) {  // all_conv2: [160][320]
    int j = i - (8192 + 32768 + 143360);
    W2ah[j] = (_Float16)WA2[j];
  }
}

// ---------------------------------------------------------------------------
// Kernel 2: one workgroup per anchor. Full GEMM chain with WMMA (quad
// accumulator chains, double-buffered fragments) + softmax weighting + heads.
// ---------------------------------------------------------------------------
__global__ __launch_bounds__(256, 2) void keynet_anchor(
    const float* __restrict__ x, const float* __restrict__ anchor,
    const float* __restrict__ W1, const float* __restrict__ b1,
    const float* __restrict__ b2,
    const _Float16* __restrict__ W2h, const _Float16* __restrict__ W5h,
    const _Float16* __restrict__ W1h, const _Float16* __restrict__ W2ah,
    const float* __restrict__ c5e, const float* __restrict__ c1e,
    const float* __restrict__ bA2,
    const float* __restrict__ Wkp1, const float* __restrict__ bkp1,
    const float* __restrict__ Wkp2, const float* __restrict__ bkp2,
    const float* __restrict__ Wat1, const float* __restrict__ bat1,
    const float* __restrict__ Wat2, const float* __restrict__ bat2,
    float* __restrict__ kpWs, float* __restrict__ attWs) {
  extern __shared__ char smem[];
  _Float16* act  = (_Float16*)smem;               // [MT][448]  x1|x2|x5
  _Float16* f1s  = act + MT * 448;                // [MT][320]
  float* wbuf = (float*)(f1s + MT * 320);         // [1024] softmax weights
  float* feat = wbuf + 1024;                      // [160]
  float* red  = feat + 160;                       // [256]
  float* h1   = red + 256;                        // [96]
  float* h2   = h1 + 96;                          // [96]
  float* xt   = h2 + 96;                          // [192] x tile

  const int a    = blockIdx.x;
  const int tid  = threadIdx.x;
  const int wave = tid >> 5;
  const int lane = tid & 31;
  const int lm   = lane & 15;
  const int off8 = (lane & 16) ? 8 : 0;   // K/row offset for lane upper half

  const float anc0 = anchor[a * 3 + 0];
  const float anc1 = anchor[a * 3 + 1];
  const float anc2 = anchor[a * 3 + 2];

  if (tid < 160) feat[tid] = 0.f;

  // ---- softmax over points of -||x - anc|| -------------------------------
  float lmax = -3.0e38f;
  for (int p = tid; p < NUM_P; p += 256) {
    float dx = x[p * 3 + 0] - anc0;
    float dy = x[p * 3 + 1] - anc1;
    float dz = x[p * 3 + 2] - anc2;
    float v = -sqrtf(dx * dx + dy * dy + dz * dz);
    wbuf[p] = v;
    lmax = fmaxf(lmax, v);
  }
  red[tid] = lmax;
  __syncthreads();
  for (int s = 128; s > 0; s >>= 1) {
    if (tid < s) red[tid] = fmaxf(red[tid], red[tid + s]);
    __syncthreads();
  }
  const float mx = red[0];
  __syncthreads();
  float lsum = 0.f;
  for (int p = tid; p < NUM_P; p += 256) {
    float e = __expf(wbuf[p] - mx);
    wbuf[p] = e;
    lsum += e;
  }
  red[tid] = lsum;
  __syncthreads();
  for (int s = 128; s > 0; s >>= 1) {
    if (tid < s) red[tid] += red[tid + s];
    __syncthreads();
  }
  const float winv = 1.f / red[0];
  __syncthreads();
  for (int p = tid; p < NUM_P; p += 256) wbuf[p] *= winv;
  __syncthreads();

  // ---- main loop over 16 point tiles of 64 -------------------------------
  for (int pt = 0; pt < NUM_P / MT; ++pt) {
    const int p0 = pt * MT;
    if (tid < MT * 3) xt[tid] = x[p0 * 3 + tid];
    __syncthreads();

    // Stage A: x1 = relu(conv1(x_rel))  (K=3, VALU) -> act[:, 0:64]
    for (int i = tid; i < MT * 64; i += 256) {
      int m = i >> 6, n = i & 63;
      float vx = xt[m * 3 + 0] - anc0;
      float vy = xt[m * 3 + 1] - anc1;
      float vz = xt[m * 3 + 2] - anc2;
      float s = b1[n] + W1[n * 3] * vx + W1[n * 3 + 1] * vy + W1[n * 3 + 2] * vz;
      act[m * 448 + n] = (_Float16)fmaxf(s, 0.f);
    }
    __syncthreads();

    // Stage B: x2 = relu(x1 @ W2^T + b2)  M=64 N=128 K=64 -> act[:, 64:192]
    for (int nt = wave; nt < 8; nt += NW) {
      v8f a0 = {}, a1 = {}, a2 = {}, a3 = {};
      gemm4<2, 448>(act + lm * 448 + off8,
                    W2h + (nt * 16 + lm) * 64 + off8, a0, a1, a2, a3);
      int n0 = nt * 16 + lm;
      float bb = b2[n0];
#pragma unroll
      for (int r = 0; r < 8; ++r) {
        int m = off8 + r;
        act[(m)      * 448 + 64 + n0] = (_Float16)fmaxf(a0[r] + bb, 0.f);
        act[(m + 16) * 448 + 64 + n0] = (_Float16)fmaxf(a1[r] + bb, 0.f);
        act[(m + 32) * 448 + 64 + n0] = (_Float16)fmaxf(a2[r] + bb, 0.f);
        act[(m + 48) * 448 + 64 + n0] = (_Float16)fmaxf(a3[r] + bb, 0.f);
      }
    }
    __syncthreads();

    // Stage C: x5 = relu(x2 @ W5a^T + c5e)  M=64 N=256 K=128 -> act[:, 192:448]
    for (int nt = wave; nt < 16; nt += NW) {
      v8f a0 = {}, a1 = {}, a2 = {}, a3 = {};
      gemm4<4, 448>(act + lm * 448 + 64 + off8,
                    W5h + (nt * 16 + lm) * 128 + off8, a0, a1, a2, a3);
      int n0 = nt * 16 + lm;
#pragma unroll
      for (int r = 0; r < 8; ++r) {
        int m = off8 + r;
        float v0 = a0[r] + c5e[(p0 + m)      * 256 + n0];
        float v1 = a1[r] + c5e[(p0 + m + 16) * 256 + n0];
        float v2 = a2[r] + c5e[(p0 + m + 32) * 256 + n0];
        float v3 = a3[r] + c5e[(p0 + m + 48) * 256 + n0];
        act[(m)      * 448 + 192 + n0] = (_Float16)fmaxf(v0, 0.f);
        act[(m + 16) * 448 + 192 + n0] = (_Float16)fmaxf(v1, 0.f);
        act[(m + 32) * 448 + 192 + n0] = (_Float16)fmaxf(v2, 0.f);
        act[(m + 48) * 448 + 192 + n0] = (_Float16)fmaxf(v3, 0.f);
      }
    }
    __syncthreads();

    // Stage D: f1 = lrelu([x1|x2|x5] @ W1r^T + c1e)  M=64 N=320 K=448
    for (int nt = wave; nt < 20; nt += NW) {
      v8f a0 = {}, a1 = {}, a2 = {}, a3 = {};
      gemm4<14, 448>(act + lm * 448 + off8,
                     W1h + (nt * 16 + lm) * 448 + off8, a0, a1, a2, a3);
      int n0 = nt * 16 + lm;
#pragma unroll
      for (int r = 0; r < 8; ++r) {
        int m = off8 + r;
        float v0 = a0[r] + c1e[(p0 + m)      * 320 + n0];
        float v1 = a1[r] + c1e[(p0 + m + 16) * 320 + n0];
        float v2 = a2[r] + c1e[(p0 + m + 32) * 320 + n0];
        float v3 = a3[r] + c1e[(p0 + m + 48) * 320 + n0];
        f1s[(m)      * 320 + n0] = (_Float16)lrelu(v0);
        f1s[(m + 16) * 320 + n0] = (_Float16)lrelu(v1);
        f1s[(m + 32) * 320 + n0] = (_Float16)lrelu(v2);
        f1s[(m + 48) * 320 + n0] = (_Float16)lrelu(v3);
      }
    }
    __syncthreads();

    // Stage E: f = f1 @ W2a^T + b ; feat += f^T * w  M=64 N=160 K=320
    for (int nt = wave; nt < 10; nt += NW) {
      v8f a0 = {}, a1 = {}, a2 = {}, a3 = {};
      gemm4<10, 320>(f1s + lm * 320 + off8,
                     W2ah + (nt * 16 + lm) * 320 + off8, a0, a1, a2, a3);
      int n0 = nt * 16 + lm;
      float bb = bA2[n0];
      float s = 0.f;
#pragma unroll
      for (int r = 0; r < 8; ++r) {
        int m = off8 + r;
        s += (a0[r] + bb) * wbuf[p0 + m];
        s += (a1[r] + bb) * wbuf[p0 + m + 16];
        s += (a2[r] + bb) * wbuf[p0 + m + 32];
        s += (a3[r] + bb) * wbuf[p0 + m + 48];
      }
      atomicAdd(&feat[n0], s);
    }
    __syncthreads();
  }

  // ---- heads (tiny GEMVs, VALU) ------------------------------------------
  if (tid < 90) {
    float s = bkp1[tid];
    for (int c = 0; c < 160; ++c) s += Wkp1[tid * 160 + c] * feat[c];
    h1[tid] = lrelu(s);
  } else if (tid >= 128 && tid < 218) {
    int j = tid - 128;
    float s = bat1[j];
    for (int c = 0; c < 160; ++c) s += Wat1[j * 160 + c] * feat[c];
    h2[j] = lrelu(s);
  }
  __syncthreads();
  if (tid < 24) {
    float s = bkp2[tid];
    for (int j = 0; j < 90; ++j) s += Wkp2[tid * 90 + j] * h1[j];
    int d = tid % 3;
    float av = (d == 0) ? anc0 : (d == 1) ? anc1 : anc2;
    kpWs[a * 24 + tid] = s + av;
  }
  if (tid == 64) {
    float s = bat2[0];
    for (int j = 0; j < 90; ++j) s += Wat2[j] * h2[j];
    attWs[a] = s;
  }
}

// ---------------------------------------------------------------------------
// Kernel 3: final single-block postprocess (scale, att softmax, argmin, kp).
// Output layout: [0:24] all_kp_x, [24:399] output_anchor, [399:524] att_x.
// ---------------------------------------------------------------------------
__global__ __launch_bounds__(128) void keynet_final(
    const float* __restrict__ anchor, const float* __restrict__ scale,
    const float* __restrict__ gt, const float* __restrict__ kpWs,
    const float* __restrict__ attWs, float* __restrict__ out) {
  __shared__ float av[128];
  __shared__ float redv[128];
  __shared__ int   redi[128];
  const int tid = threadIdx.x;
  const float s0 = scale[0], s1 = scale[1], s2 = scale[2];

  for (int i = tid; i < 375; i += 128) {
    int d = i % 3;
    out[24 + i] = anchor[i] * ((d == 0) ? s0 : (d == 1) ? s1 : s2);
  }

  float v = (tid < NUM_A) ? attWs[tid] : -3.0e38f;
  av[tid] = v;
  redv[tid] = v;
  __syncthreads();
  for (int s = 64; s > 0; s >>= 1) {
    if (tid < s) redv[tid] = fmaxf(redv[tid], redv[tid + s]);
    __syncthreads();
  }
  const float mx = redv[0];
  __syncthreads();
  float e = (tid < NUM_A) ? __expf(av[tid] - mx) : 0.f;
  redv[tid] = e;
  __syncthreads();
  for (int s = 64; s > 0; s >>= 1) {
    if (tid < s) redv[tid] += redv[tid + s];
    __syncthreads();
  }
  const float inv = 1.f / redv[0];
  if (tid < NUM_A) out[399 + tid] = e * inv;
  __syncthreads();

  float d2 = 3.0e38f;
  if (tid < NUM_A) {
    float dx = anchor[tid * 3 + 0] * s0 - gt[0];
    float dy = anchor[tid * 3 + 1] * s1 - gt[1];
    float dz = anchor[tid * 3 + 2] * s2 - gt[2];
    d2 = dx * dx + dy * dy + dz * dz;
  }
  redv[tid] = d2;
  redi[tid] = tid;
  __syncthreads();
  for (int s = 64; s > 0; s >>= 1) {
    if (tid < s) {
      float ov = redv[tid + s];
      int oi = redi[tid + s];
      if (ov < redv[tid] || (ov == redv[tid] && oi < redi[tid])) {
        redv[tid] = ov;
        redi[tid] = oi;
      }
    }
    __syncthreads();
  }
  const int mi = redi[0];
  if (tid < 24) {
    int d = tid % 3;
    out[tid] = kpWs[mi * 24 + tid] * ((d == 0) ? s0 : (d == 1) ? s1 : s2);
  }
}

// ---------------------------------------------------------------------------
extern "C" void kernel_launch(void* const* d_in, const int* in_sizes, int n_in,
                              void* d_out, int out_size, void* d_ws, size_t ws_size,
                              hipStream_t stream) {
  (void)in_sizes; (void)n_in; (void)out_size; (void)ws_size;
  const float* out_img = (const float*)d_in[0];
  const int*   choose  = (const int*)d_in[1];
  const float* x       = (const float*)d_in[2];
  const float* anchor  = (const float*)d_in[3];
  const float* scale   = (const float*)d_in[4];
  const float* gt      = (const float*)d_in[5];
  const float* W1  = (const float*)d_in[6];  const float* b1  = (const float*)d_in[7];
  const float* W2  = (const float*)d_in[8];  const float* b2  = (const float*)d_in[9];
  const float* We1 = (const float*)d_in[10]; const float* be1 = (const float*)d_in[11];
  const float* We2 = (const float*)d_in[12]; const float* be2 = (const float*)d_in[13];
  const float* W5  = (const float*)d_in[14]; const float* b5  = (const float*)d_in[15];
  const float* WA1 = (const float*)d_in[16]; const float* bA1 = (const float*)d_in[17];
  const float* WA2 = (const float*)d_in[18]; const float* bA2 = (const float*)d_in[19];
  const float* Wkp1 = (const float*)d_in[20]; const float* bkp1 = (const float*)d_in[21];
  const float* Wkp2 = (const float*)d_in[22]; const float* bkp2 = (const float*)d_in[23];
  const float* Wat1 = (const float*)d_in[24]; const float* bat1 = (const float*)d_in[25];
  const float* Wat2 = (const float*)d_in[26]; const float* bat2 = (const float*)d_in[27];

  char* ws = (char*)d_ws;
  float*    c5e  = (float*)(ws + 0);            // 1024*256*4 = 1,048,576
  float*    c1e  = (float*)(ws + 1048576);      // 1024*320*4 = 1,310,720
  _Float16* W2h  = (_Float16*)(ws + 2359296);   // 8192*2
  _Float16* W5h  = (_Float16*)(ws + 2375680);   // 32768*2
  _Float16* W1h  = (_Float16*)(ws + 2441216);   // 143360*2
  _Float16* W2ah = (_Float16*)(ws + 2727936);   // 51200*2
  float*    kpWs = (float*)(ws + 2830336);      // 125*24*4
  float*    attWs= (float*)(ws + 2842336);      // 125*4
  float*    out  = (float*)d_out;

  keynet_point<<<NUM_P, 128, 0, stream>>>(out_img, choose, We1, be1, We2, be2,
                                          W5, b5, WA1, bA1, c5e, c1e);
  keynet_wconv<<<920, 256, 0, stream>>>(W2, W5, WA1, WA2, W2h, W5h, W1h, W2ah);

  const int smemBytes = (MT * 448 + MT * 320) * 2 +
                        (1024 + 160 + 256 + 96 + 96 + 192) * 4;  // 105,600 B
  keynet_anchor<<<NUM_A, 256, smemBytes, stream>>>(
      x, anchor, W1, b1, b2, W2h, W5h, W1h, W2ah, c5e, c1e, bA2,
      Wkp1, bkp1, Wkp2, bkp2, Wat1, bat1, Wat2, bat2, kpWs, attWs);

  keynet_final<<<1, 128, 0, stream>>>(anchor, scale, gt, kpWs, attWs, out);
}
